// BitNetMLP_4887672783069
// MI455X (gfx1250) — 8-bit matrix operand order repaired and checked by reading, compiled, not run on hardware
//
#include <hip/hip_runtime.h>
#include <hip/hip_bf16.h>
#include <math.h>

typedef __attribute__((ext_vector_type(8))) int v8i;

#define EPS_Q 1e-5f

constexpr int H = 2048;
constexpr int I = 8192;
constexpr int T = 4 * 4096;

__device__ __forceinline__ int clamp_i(int q, int lo, int hi) {
    return q < lo ? lo : (q > hi ? hi : q);
}

__device__ __forceinline__ void async_ldscpy32(unsigned lds_off, const void* gptr) {
    unsigned long long ga = (unsigned long long)(uintptr_t)gptr;
    asm volatile("global_load_async_to_lds_b128 %0, %1, off\n\t"
                 "global_load_async_to_lds_b128 %0, %1, off offset:16"
                 :: "v"(lds_off), "v"(ga) : "memory");
}

__device__ __forceinline__ void wait_async0() {
    asm volatile("s_wait_asynccnt 0x0" ::: "memory");
}

__device__ __forceinline__ unsigned lds_offset_of(const void* p) {
    return (unsigned)(uintptr_t)p;
}

__global__ void bitnet_absum_partial(const float* __restrict__ w, long long n,
                                     float* __restrict__ partial) {
    __shared__ float red[256];
    long long idx = (long long)blockIdx.x * blockDim.x + threadIdx.x;
    long long stride = (long long)gridDim.x * blockDim.x;
    float s = 0.0f;
    for (long long i = idx; i < n; i += stride) s += fabsf(w[i]);
    red[threadIdx.x] = s;
    __syncthreads();
    for (int o = 128; o > 0; o >>= 1) {
        if (threadIdx.x < o) red[threadIdx.x] += red[threadIdx.x + o];
        __syncthreads();
    }
    if (threadIdx.x == 0) partial[blockIdx.x] = red[0];
}

__global__ void bitnet_finalize_scale(const float* __restrict__ partial, int nparts,
                                      long long n, float* __restrict__ scale_out) {
    __shared__ float red[256];
    float s = 0.0f;
    for (int i = threadIdx.x; i < nparts; i += 256) s += partial[i];
    red[threadIdx.x] = s;
    __syncthreads();
    for (int o = 128; o > 0; o >>= 1) {
        if (threadIdx.x < o) red[threadIdx.x] += red[threadIdx.x + o];
        __syncthreads();
    }
    if (threadIdx.x == 0) *scale_out = red[0] / (float)n + EPS_Q;
}

__global__ void bitnet_quant_weight_packed(const float* __restrict__ w, int Kdim,
                                           long long ndwords,
                                           const float* __restrict__ scale_p,
                                           int* __restrict__ wq) {
    const float inv = 1.0f / (*scale_p);
    const int ktiles = Kdim >> 6;
    long long idx = (long long)blockIdx.x * blockDim.x + threadIdx.x;
    long long stride = (long long)gridDim.x * blockDim.x;
    for (long long d = idx; d < ndwords; d += stride) {
        const int r     = (int)(d & 255);
        const long long tile = d >> 8;
        const int lane  = r >> 3;
        const int j     = r & 7;
        const int ntile = (int)(tile / ktiles);
        const int ktile = (int)(tile % ktiles);
        const int k  = (ktile << 6) + ((j >> 1) << 4) + ((j & 1) << 2) + ((lane >> 4) << 3);
        const int n0 = (ntile << 4) + (lane & 15);
        unsigned int packed = 0;
#pragma unroll
        for (int b = 0; b < 4; ++b) {
            float wv = w[(size_t)n0 * Kdim + k + b];
            int q = clamp_i((int)rintf(wv * inv), -1, 1);
            packed |= ((unsigned int)(q & 0xff)) << (8 * b);
        }
        wq[d] = (int)packed;
    }
}

__global__ void bitnet_rowmax(const float* __restrict__ x, int D,
                              float* __restrict__ xs) {
    __shared__ float red[256];
    const int t = blockIdx.x;
    const float* row = x + (size_t)t * D;
    float m = 0.0f;
    for (int i = threadIdx.x; i < D; i += 256) m = fmaxf(m, fabsf(row[i]));
    red[threadIdx.x] = m;
    __syncthreads();
    for (int o = 128; o > 0; o >>= 1) {
        if (threadIdx.x < o) red[threadIdx.x] = fmaxf(red[threadIdx.x], red[threadIdx.x + o]);
        __syncthreads();
    }
    if (threadIdx.x == 0) xs[t] = red[0] / 127.0f + EPS_Q;
}

__global__ void bitnet_pack_act(const float* __restrict__ x,
                                const float* __restrict__ xs, int D,
                                long long ndwords, int* __restrict__ xq) {
    const int ktiles = D >> 6;
    long long idx = (long long)blockIdx.x * blockDim.x + threadIdx.x;
    long long stride = (long long)gridDim.x * blockDim.x;
    for (long long d = idx; d < ndwords; d += stride) {
        const int r     = (int)(d & 255);
        const long long tile = d >> 8;
        const int lane  = r >> 3;
        const int j     = r & 7;
        const int mtile = (int)(tile / ktiles);
        const int ktile = (int)(tile % ktiles);
        const int m = lane & 15, g = lane >> 4;
        const int t = (mtile << 4) + m;
        const int k = (ktile << 6) + ((j >> 1) << 4) + ((j & 1) << 2) + (g << 3);
        const float4 v = *(const float4*)(x + (size_t)t * D + k);
        const float inv = 1.0f / xs[t];
        int q0 = clamp_i((int)rintf(v.x * inv), -128, 127);
        int q1 = clamp_i((int)rintf(v.y * inv), -128, 127);
        int q2 = clamp_i((int)rintf(v.z * inv), -128, 127);
        int q3 = clamp_i((int)rintf(v.w * inv), -128, 127);
        unsigned int packed = (unsigned int)(q0 & 0xff) |
                              ((unsigned int)(q1 & 0xff) << 8) |
                              ((unsigned int)(q2 & 0xff) << 16) |
                              ((unsigned int)(q3 & 0xff) << 24);
        xq[d] = (int)packed;
    }
}

template <bool FUSED>
__global__ __launch_bounds__(256, 1)
void bitnet_gemm_iu8(const int* __restrict__ Ap,
                     const float* __restrict__ As,
                     const int* __restrict__ B0p,
                     const int* __restrict__ B1p,
                     const float* __restrict__ wscale0,
                     const float* __restrict__ wscale1,
                     int Kdim, int Ntotal,
                     float* __restrict__ Out)
{
    __shared__ __align__(32) int lds_a[2][8 * 256];
    __shared__ __align__(32) int lds_b0[2][4 * 256];
    __shared__ __align__(32) int lds_b1[2][4 * 256];

    const int tid  = threadIdx.x;
    const int lane = tid & 31;
    const int wave = tid >> 5;
    const int wm   = wave & 3;
    const int wn   = wave >> 2;
    const int ktiles = Kdim >> 6;
    const int mtile0 = blockIdx.x * 8;
    const int ntile0 = blockIdx.y * 4;

    v8i accg[2][2] = {};
    v8i accu[2][2] = {};

    const int sc = tid >> 5;
    const int so = tid & 31;
    const int bt = tid & 127;
    const int bc = bt >> 5;
    const int bo = bt & 31;

    const int* myB  = (tid < 128) ? B0p : B1p;
    const bool doB  = FUSED || (tid < 128);

    auto stage = [&](int kt, int buf) {
        unsigned la = lds_offset_of(&lds_a[buf][sc * 256]) + so * 32;
        const char* ga = (const char*)(Ap + ((size_t)(mtile0 + sc) * ktiles + kt) * 256)
                         + so * 32;
        async_ldscpy32(la, ga);
        if (doB) {
            int* dstbase = (tid < 128) ? lds_b0[buf] : lds_b1[buf];
            unsigned lb = lds_offset_of(dstbase + bc * 256) + bo * 32;
            const char* gb = (const char*)(myB + ((size_t)(ntile0 + bc) * ktiles + kt) * 256)
                             + bo * 32;
            async_ldscpy32(lb, gb);
        }
    };

    stage(0, 0);

    for (int kt = 0; kt < ktiles; ++kt) {
        const int cur = kt & 1;
        wait_async0();
        __syncthreads();
        if (kt + 1 < ktiles) stage(kt + 1, cur ^ 1);

        const v8i* aF  = (const v8i*)lds_a[cur];
        const v8i* b0F = (const v8i*)lds_b0[cur];
        const v8i* b1F = (const v8i*)lds_b1[cur];
        v8i afrag[2], b0frag[2], b1frag[2];
#pragma unroll
        for (int i = 0; i < 2; ++i) {
            afrag[i]  = aF[(wm * 2 + i) * 32 + lane];
            b0frag[i] = b0F[(wn * 2 + i) * 32 + lane];
            if (FUSED) b1frag[i] = b1F[(wn * 2 + i) * 32 + lane];
        }

#pragma unroll
        for (int mi = 0; mi < 2; ++mi) {
#pragma unroll
            for (int nj = 0; nj < 2; ++nj) {
                accg[mi][nj] = __builtin_amdgcn_wmma_i32_16x16x64_iu8(
                    true, afrag[mi], true, b0frag[nj], accg[mi][nj], false, false);
                if (FUSED)
                    accu[mi][nj] = __builtin_amdgcn_wmma_i32_16x16x64_iu8(
                        true, afrag[mi], true, b1frag[nj], accu[mi][nj], false, false);
            }
        }
    }

    const float s0 = *wscale0;
    float s1 = 0.0f;
    if constexpr (FUSED) s1 = *wscale1;

    const int rowbase = blockIdx.x * 128;
    const int colbase = blockIdx.y * 64;
#pragma unroll
    for (int mi = 0; mi < 2; ++mi) {
#pragma unroll
        for (int nj = 0; nj < 2; ++nj) {
#pragma unroll
            for (int v = 0; v < 8; ++v) {
                const int M = (wm * 2 + mi) * 16 + v + (lane >> 4) * 8;
                const int N = (wn * 2 + nj) * 16 + (lane & 15);
                const size_t token = (size_t)(rowbase + M);
                const float xsv = As[token];
                const float y0 = (float)accg[mi][nj][v] * xsv * s0;
                float res;
                if constexpr (FUSED) {
                    const float y1 = (float)accu[mi][nj][v] * xsv * s1;
                    res = (y0 / (1.0f + __expf(-y0))) * y1;
                } else {
                    res = y0;
                }
                Out[token * (size_t)Ntotal + (colbase + N)] = res;
            }
        }
    }
}

extern "C" void kernel_launch(void* const* d_in, const int* in_sizes, int n_in,
                              void* d_out, int out_size, void* d_ws, size_t ws_size,
                              hipStream_t stream) {
    const float* x      = (const float*)d_in[0];
    const float* w_gate = (const float*)d_in[1];
    const float* w_up   = (const float*)d_in[2];
    const float* w_down = (const float*)d_in[3];
    float* out = (float*)d_out;

    char* p = (char*)d_ws;
    auto alloc = [&](size_t bytes) {
        char* r = p;
        p += (bytes + 255) & ~(size_t)255;
        return r;
    };
    float* partials = (float*)alloc(3 * 1024 * sizeof(float));
    float* scales   = (float*)alloc(4 * sizeof(float));
    float* xs       = (float*)alloc((size_t)T * sizeof(float));
    float* hs       = (float*)alloc((size_t)T * sizeof(float));
    int* wqg        = (int*)alloc((size_t)I * H);
    int* wqu        = (int*)alloc((size_t)I * H);
    int* wqd        = (int*)alloc((size_t)H * I);
    int* xq         = (int*)alloc((size_t)T * H);
    int* hq         = (int*)alloc((size_t)T * I);
    float* hbuf     = (float*)alloc((size_t)T * I * sizeof(float));

    const long long NW  = (long long)I * H;
    const long long NWD = NW / 4;
    const long long NXD = (long long)T * H / 4;
    const long long NHD = (long long)T * I / 4;

    bitnet_absum_partial<<<1024, 256, 0, stream>>>(w_gate, NW, partials + 0);
    bitnet_absum_partial<<<1024, 256, 0, stream>>>(w_up,   NW, partials + 1024);
    bitnet_absum_partial<<<1024, 256, 0, stream>>>(w_down, NW, partials + 2048);
    bitnet_finalize_scale<<<1, 256, 0, stream>>>(partials + 0,    1024, NW, scales + 0);
    bitnet_finalize_scale<<<1, 256, 0, stream>>>(partials + 1024, 1024, NW, scales + 1);
    bitnet_finalize_scale<<<1, 256, 0, stream>>>(partials + 2048, 1024, NW, scales + 2);

    bitnet_quant_weight_packed<<<4096, 256, 0, stream>>>(w_gate, H, NWD, scales + 0, wqg);
    bitnet_quant_weight_packed<<<4096, 256, 0, stream>>>(w_up,   H, NWD, scales + 1, wqu);
    bitnet_quant_weight_packed<<<4096, 256, 0, stream>>>(w_down, I, NWD, scales + 2, wqd);

    bitnet_rowmax<<<T, 256, 0, stream>>>(x, H, xs);
    bitnet_pack_act<<<4096, 256, 0, stream>>>(x, xs, H, NXD, xq);

    dim3 g1(T / 128, I / 64);
    bitnet_gemm_iu8<true><<<g1, 256, 0, stream>>>(xq, xs, wqg, wqu,
                                                  scales + 0, scales + 1,
                                                  H, I, hbuf);

    bitnet_rowmax<<<T, 256, 0, stream>>>(hbuf, I, hs);
    bitnet_pack_act<<<8192, 256, 0, stream>>>(hbuf, hs, I, NHD, hq);

    dim3 g2(T / 128, H / 64);
    bitnet_gemm_iu8<false><<<g2, 256, 0, stream>>>(hq, hs, wqd, wqd,
                                                   scales + 2, scales + 2,
                                                   I, H, out);
}
